// VQVAE_1486058684748
// MI455X (gfx1250) — compile-verified, matrix operand order checked by reading, not run on hardware
//
#include <hip/hip_runtime.h>
#include <math.h>

// ---------------------------------------------------------------------------
// VQ-VAE forward loss, fused per-image, gfx1250 (MI455X, wave32, WMMA).
// One block = one image (B = 4096). All activations live in LDS (~54 KB).
// GEMM stages on v_wmma_f32_16x16x32_bf16 (K-loops unrolled for XDL/VALU
// overlap); input/codebook staged via GLOBAL_LOAD_ASYNC_TO_LDS_B128
// (ASYNCcnt). All fragment gathers are branchless (clamped index + select)
// so EXEC is never diverged around WMMA.
// ---------------------------------------------------------------------------

typedef __attribute__((ext_vector_type(16))) __bf16 v16bf;
typedef __attribute__((ext_vector_type(8)))  float  v8f;
typedef int v4i __attribute__((__vector_size__(4 * sizeof(int))));

#define NIMG  4096
#define BETA  0.1f

#define AS1 __attribute__((address_space(1)))
#define AS3 __attribute__((address_space(3)))

#if defined(__AMDGCN__) && __has_builtin(__builtin_amdgcn_global_load_async_to_lds_b32)
#define HAVE_ASYNC_LDS 1
#else
#define HAVE_ASYNC_LDS 0
#endif
#if defined(__AMDGCN__) && __has_builtin(__builtin_amdgcn_global_load_async_to_lds_b128)
#define HAVE_ASYNC_B128 1
#else
#define HAVE_ASYNC_B128 0
#endif

// Copy 16 bytes (4 floats) global -> LDS, async when available.
__device__ __forceinline__ void async_16B_to_lds(const float* g, float* l) {
#if HAVE_ASYNC_B128
    __builtin_amdgcn_global_load_async_to_lds_b128((AS1 v4i*)g, (AS3 v4i*)l, 0, 0);
#elif HAVE_ASYNC_LDS
    __builtin_amdgcn_global_load_async_to_lds_b32((AS1 int*)g, (AS3 int*)l, 0,  0);
    __builtin_amdgcn_global_load_async_to_lds_b32((AS1 int*)g, (AS3 int*)l, 4,  0);
    __builtin_amdgcn_global_load_async_to_lds_b32((AS1 int*)g, (AS3 int*)l, 8,  0);
    __builtin_amdgcn_global_load_async_to_lds_b32((AS1 int*)g, (AS3 int*)l, 12, 0);
#else
    l[0] = g[0]; l[1] = g[1]; l[2] = g[2]; l[3] = g[3];
#endif
}

__device__ __forceinline__ void wait_async_lds() {
#if HAVE_ASYNC_LDS || HAVE_ASYNC_B128
#if __has_builtin(__builtin_amdgcn_s_wait_asynccnt)
    __builtin_amdgcn_s_wait_asynccnt(0);
#else
    asm volatile("s_wait_asynccnt 0" ::: "memory");
#endif
#endif
}

__device__ __forceinline__ float  bf2f(__bf16 v) { return (float)v; }
__device__ __forceinline__ __bf16 f2bf(float v)  { return (__bf16)v; }

// Fragment K index for 16-bit A/B operands of v_wmma_*_16x16x32_* (ISA 7.12.2):
// lane half (lane>>4) selects K block; slots 0..7 -> K = half*8+j,
// slots 8..15 -> K = 16 + half*8 + (j-8).
__device__ __forceinline__ int frag_k(int half, int j) {
    return (j < 8) ? (half * 8 + j) : (16 + half * 8 + (j - 8));
}

struct __align__(16) SmemU {
    union {
        float  embf[64 * 64];      // staged codebook f32 (stage 0)    16 KB
        __bf16 h1[32 * 16 * 16];   // conv1 output (c, y*16+x)         16 KB
        float  scores[64 * 64];    // VQ scores (t, k)                 16 KB
        __bf16 dbuf[16 * 16 * 32]; // convT1 output (y, x, c)          16 KB
    } u;
};

struct __align__(16) Smem {
    float  x[3 * 32 * 32];         // input image (c,y,x)              12 KB
    SmemU  un;                     //                                  16 KB
    float  xenc[64 * 64];          // encoder tokens (t,d) f32         16 KB
    __bf16 embb[64 * 64];          // codebook bf16 (k,d)               8 KB
    float  c2[64];                 // ||e_k||^2
    int    tokens[64];
    float  red[256];
};

__global__ __launch_bounds__(256)
void vqvae_fused_kernel(const float* __restrict__ x,
                        const float* __restrict__ w1,  const float* __restrict__ b1,
                        const float* __restrict__ w2,  const float* __restrict__ b2,
                        const float* __restrict__ wd1, const float* __restrict__ bd1,
                        const float* __restrict__ wd2, const float* __restrict__ bd2,
                        const float* __restrict__ emb,
                        float* __restrict__ out)
{
    __shared__ Smem s;
    const int tid  = threadIdx.x;
    const int lane = tid & 31;
    const int half = lane >> 4;
    const int wave = tid >> 5;
    const int img  = blockIdx.x;

    // L2-resident weights: prefetch hint (global_prefetch_b8).
    __builtin_prefetch(w2  + tid, 0, 1);
    __builtin_prefetch(wd1 + tid, 0, 1);

    // ---- Stage 0: async-stage image (f32) + codebook (f32) into LDS --------
    const float* ximg = x + (size_t)img * (3 * 32 * 32);
#pragma unroll
    for (int c = 0; c < 3; ++c) {              // 768 16B-chunks of x
        const int i = (c * 256 + tid) * 4;
        async_16B_to_lds(ximg + i, &s.x[i]);
    }
#pragma unroll
    for (int c = 0; c < 4; ++c) {              // 1024 16B-chunks of emb
        const int i = (c * 256 + tid) * 4;
        async_16B_to_lds(emb + i, &s.un.u.embf[i]);
    }
    wait_async_lds();
    __syncthreads();

    // codebook: bf16 copy + ||e_k||^2 from the staged f32 image of emb
    for (int i = tid; i < 64 * 64; i += 256) s.embb[i] = f2bf(s.un.u.embf[i]);
    if (tid < 64) {
        float acc = 0.f;
        for (int d = 0; d < 64; ++d) { float v = s.un.u.embf[tid * 64 + d]; acc += v * v; }
        s.c2[tid] = acc;
    }
    __syncthreads();   // union region now free for h1

    // ---- Stage 1: conv1 3->32, k3 s2 p1, ReLU ------------------------------
    // Implicit GEMM: M=256 output pixels (16x16), N=32 ch, K=27 padded to 32.
    for (int job = wave; job < 32; job += 8) {
        const int mt = job >> 1, nt = job & 1;
        const int pix = mt * 16 + (lane & 15);
        const int oy = pix >> 4, ox = pix & 15;
        const int nch = nt * 16 + (lane & 15);
        v16bf a, b;
#pragma unroll
        for (int j = 0; j < 16; ++j) {
            const int k = frag_k(half, j);
            const int kc = k < 27 ? k : 26;          // clamp (branchless)
            const int ci = kc / 9, r = kc % 9;
            const int iy = 2 * oy + (r / 3) - 1;
            const int ix = 2 * ox + (r % 3) - 1;
            const bool vin = ((unsigned)iy < 32u) & ((unsigned)ix < 32u) & (k < 27);
            const float xv = s.x[ci * 1024 + (iy & 31) * 32 + (ix & 31)];
            const float wv = w1[nch * 27 + kc];
            a[j] = f2bf(vin ? xv : 0.f);
            b[j] = f2bf((k < 27) ? wv : 0.f);
        }
        v8f acc = {};
        acc = __builtin_amdgcn_wmma_f32_16x16x32_bf16(false, a, false, b,
                                                      (short)0, acc, false, false);
        const int ncol = nt * 16 + (lane & 15);
#pragma unroll
        for (int r = 0; r < 8; ++r) {
            const int mm = mt * 16 + half * 8 + r;
            float v = acc[r] + b1[ncol];
            v = v > 0.f ? v : 0.f;
            s.un.u.h1[ncol * 256 + mm] = f2bf(v);    // (c, y*16+x)
        }
    }
    __syncthreads();

    // ---- Stage 2: conv2 32->64, k3 s2 p1 (no ReLU) -------------------------
    // M = 64 tokens (t = x*8+y), N = 64 ch, K = 288 (9 WMMA steps, unrolled).
    for (int job = wave; job < 16; job += 8) {
        const int mt = job >> 2, nt = job & 3;
        const int t  = mt * 16 + (lane & 15);
        const int oy = t & 7, ox = t >> 3;           // token t -> pixel (y,x)
        const int nch = nt * 16 + (lane & 15);
        v8f acc = {};
#pragma unroll
        for (int ks = 0; ks < 9; ++ks) {
            v16bf a, b;
#pragma unroll
            for (int j = 0; j < 16; ++j) {
                const int K = ks * 32 + frag_k(half, j);
                const int ci = K / 9, r = K % 9;
                const int iy = 2 * oy + (r / 3) - 1;
                const int ix = 2 * ox + (r % 3) - 1;
                const bool vin = ((unsigned)iy < 16u) & ((unsigned)ix < 16u);
                const float hv = bf2f(s.un.u.h1[ci * 256 + (iy & 15) * 16 + (ix & 15)]);
                a[j] = f2bf(vin ? hv : 0.f);
                b[j] = f2bf(w2[nch * 288 + K]);      // (o, ci, ky, kx) flat
            }
            acc = __builtin_amdgcn_wmma_f32_16x16x32_bf16(false, a, false, b,
                                                          (short)0, acc, false, false);
        }
        const int ncol = nt * 16 + (lane & 15);
#pragma unroll
        for (int r = 0; r < 8; ++r) {
            const int tt = mt * 16 + half * 8 + r;
            s.xenc[tt * 64 + ncol] = acc[r] + b2[ncol];
        }
    }
    __syncthreads();

    // ---- Stage 3: VQ scores = ||e_k||^2 - 2 * xenc . e_k  (M=N=K=64) -------
    for (int job = wave; job < 16; job += 8) {
        const int mt = job >> 2, nt = job & 3;
        v8f acc = {};
#pragma unroll
        for (int ks = 0; ks < 2; ++ks) {
            v16bf a, b;
#pragma unroll
            for (int j = 0; j < 16; ++j) {
                const int d = ks * 32 + frag_k(half, j);
                a[j] = f2bf(s.xenc[(mt * 16 + (lane & 15)) * 64 + d]);
                b[j] = s.embb[(nt * 16 + (lane & 15)) * 64 + d];   // B[d][k]=emb[k][d]
            }
            acc = __builtin_amdgcn_wmma_f32_16x16x32_bf16(false, a, false, b,
                                                          (short)0, acc, false, false);
        }
        const int ncol = nt * 16 + (lane & 15);
#pragma unroll
        for (int r = 0; r < 8; ++r) {
            const int tt = mt * 16 + half * 8 + r;
            s.un.u.scores[tt * 64 + ncol] = s.c2[ncol] - 2.f * acc[r];
        }
    }
    __syncthreads();

    // ---- Stage 3b: argmin per token + VQ norm term -------------------------
    float local_norm = 0.f;
    if (tid < 64) {
        const float* sc = &s.un.u.scores[tid * 64];
        float best = sc[0]; int bi = 0;
        for (int k = 1; k < 64; ++k) { const float v = sc[k]; if (v < best) { best = v; bi = k; } }
        s.tokens[tid] = bi;
        float acc = 0.f;
        for (int d = 0; d < 64; ++d) {
            const float diff = emb[bi * 64 + d] - s.xenc[tid * 64 + d];
            acc += diff * diff;
        }
        local_norm = sqrtf(acc);
    }
    __syncthreads();   // scores dead; union reused as dbuf below

    // ---- Stage 4: convT1 64->32, k4 s2 p1, ReLU ----------------------------
    // torch convT: oy = 2*iy - 1 + ky. Parity decomposition: for output
    // parity p, ky = 2*tap + (1-p), iy = ((oy+1)>>1) - tap  (2x2 tap set).
    // 4 parity classes x (M=64, N=32, K=256) GEMMs -> 32 jobs.
    for (int job = wave; job < 32; job += 8) {
        const int cls = job >> 3;
        const int mt  = (job >> 1) & 3;
        const int nt  = job & 1;
        const int py = cls & 1, px = cls >> 1;
        const int m  = mt * 16 + (lane & 15);
        const int oy = 2 * (m & 7) + py, ox = 2 * (m >> 3) + px;
        const int nch = nt * 16 + (lane & 15);
        v8f acc = {};
#pragma unroll
        for (int ks = 0; ks < 8; ++ks) {
            v16bf a, b;
#pragma unroll
            for (int j = 0; j < 16; ++j) {
                const int K  = ks * 32 + frag_k(half, j);
                const int tap = K >> 6, ci = K & 63;
                const int tY = tap >> 1, tX = tap & 1;
                const int iy = ((oy + 1) >> 1) - tY;
                const int ix = ((ox + 1) >> 1) - tX;
                const bool vin = ((unsigned)iy < 8u) & ((unsigned)ix < 8u);
                const int tok = s.tokens[((ix & 7) * 8 + (iy & 7))];
                const float vv = bf2f(s.embb[tok * 64 + ci]);       // vq feature map
                a[j] = f2bf(vin ? vv : 0.f);
                const int ky = 2 * tY + (1 - py);
                const int kx = 2 * tX + (1 - px);
                b[j] = f2bf(wd1[ci * 512 + nch * 16 + ky * 4 + kx]); // (Cin,Cout,4,4)
            }
            acc = __builtin_amdgcn_wmma_f32_16x16x32_bf16(false, a, false, b,
                                                          (short)0, acc, false, false);
        }
        const int ncol = nt * 16 + (lane & 15);
#pragma unroll
        for (int r = 0; r < 8; ++r) {
            const int mm  = mt * 16 + half * 8 + r;
            const int oy2 = 2 * (mm & 7) + py, ox2 = 2 * (mm >> 3) + px;
            float v = acc[r] + bd1[ncol];
            v = v > 0.f ? v : 0.f;
            s.un.u.dbuf[(oy2 * 16 + ox2) * 32 + ncol] = f2bf(v);   // (y,x,c)
        }
    }
    __syncthreads();

    // ---- Stage 5: convT2 32->3 (f32 VALU) + recon loss ---------------------
    float local_recon = 0.f;
    for (int o = tid; o < 3 * 32 * 32; o += 256) {
        const int co = o >> 10;
        const int oy = (o >> 5) & 31, ox = o & 31;
        float acc = bd2[co];
        const int byY = (oy + 1) >> 1, pyy = oy & 1;
        const int bxX = (ox + 1) >> 1, pxx = ox & 1;
#pragma unroll
        for (int tY = 0; tY < 2; ++tY) {
            const int iy = byY - tY;
            const int ky = 2 * tY + (1 - pyy);
            const bool vy = (unsigned)iy < 16u;
#pragma unroll
            for (int tX = 0; tX < 2; ++tX) {
                const int ix = bxX - tX;
                const int kx = 2 * tX + (1 - pxx);
                const bool v = vy & ((unsigned)ix < 16u);
                const __bf16* drow = &s.un.u.dbuf[((iy & 15) * 16 + (ix & 15)) * 32];
                const float* wrow = &wd2[co * 16 + ky * 4 + kx];    // + ci*48
                float part = 0.f;
                for (int ci = 0; ci < 32; ++ci)
                    part += bf2f(drow[ci]) * wrow[ci * 48];
                acc += v ? part : 0.f;
            }
        }
        const float diff = s.x[o] - acc;
        local_recon += 0.5f * diff * diff;
    }

    // ---- Reduce + accumulate mean-over-batch loss --------------------------
    s.red[tid] = local_recon + (1.f + BETA) * local_norm;
    __syncthreads();
    for (int off = 128; off > 0; off >>= 1) {
        if (tid < off) s.red[tid] += s.red[tid + off];
        __syncthreads();
    }
    if (tid == 0) atomicAdd(out, s.red[0] * (1.f / (float)NIMG));
}

__global__ void vqvae_zero_out(float* p) { if (threadIdx.x == 0) p[0] = 0.f; }

extern "C" void kernel_launch(void* const* d_in, const int* in_sizes, int n_in,
                              void* d_out, int out_size, void* d_ws, size_t ws_size,
                              hipStream_t stream) {
    (void)in_sizes; (void)n_in; (void)out_size; (void)d_ws; (void)ws_size;
    const float* x   = (const float*)d_in[0];
    const float* w1  = (const float*)d_in[1];
    const float* b1  = (const float*)d_in[2];
    const float* w2  = (const float*)d_in[3];
    const float* b2  = (const float*)d_in[4];
    const float* wd1 = (const float*)d_in[5];
    const float* bd1 = (const float*)d_in[6];
    const float* wd2 = (const float*)d_in[7];
    const float* bd2 = (const float*)d_in[8];
    const float* emb = (const float*)d_in[9];
    float* out = (float*)d_out;

    vqvae_zero_out<<<1, 32, 0, stream>>>(out);
    vqvae_fused_kernel<<<NIMG, 256, 0, stream>>>(x, w1, b1, w2, b2,
                                                 wd1, bd1, wd2, bd2, emb, out);
}
